// RegionProposalNetwork_45603962749092
// MI455X (gfx1250) — matrix operand order repaired and checked by reading, not compiled, not run on hardware
//
#include <hip/hip_runtime.h>
#include <hip/hip_bf16.h>

typedef __attribute__((ext_vector_type(16))) __bf16 v16bf;
typedef __attribute__((ext_vector_type(8)))  float  v8f;

#define HW      128
#define CIN     512
#define COUT    512
#define K3      9
#define KTOT    (CIN * K3)          // 4608
#define MTOT    (HW * HW)           // 16384
#define NANCH   (MTOT * 9)          // 147456
#define LDSS    40                  // LDS row stride in bf16 (80B = 5*16B)
#define NT      64                  // block N tile
#define STRIP   136                 // padded A strip rows in LDS (uses 0..129)
#define PADW    136                 // padded feat row length
#define PADH    130                 // padded feat rows
#define NK3     48                  // outer K iters: 3 dy * 16 channel chunks
#define CAND_MAX 4096
#define PRE_TOPK 2000
#define OUT_TOPK 1000
#define BBOX_CLIP 4.135166556742356f
#define IMG_SZ   2048.0f

// ---------------- workspace layout (all 256B aligned) ----------------
static constexpr size_t SZ_FEATP = (size_t)CIN * PADH * PADW * 2;  // 18.1 MB padded bf16 feat
static constexpr size_t SZ_BWT   = (size_t)COUT * KTOT * 2;        // 4.5 MB  Bwt[cout][k]
static constexpr size_t SZ_RPNB  = (size_t)MTOT * COUT * 2;        // 16 MB
static constexpr size_t SZ_W2T   = (size_t)48 * 512 * 2;
static constexpr size_t SZ_B2    = 256;
static constexpr size_t SZ_OUT2  = (size_t)MTOT * 48 * 4;
static constexpr size_t SZ_SCORE = (size_t)NANCH * 4;
static constexpr size_t SZ_PROPS = (size_t)NANCH * 16;
static constexpr size_t SZ_VALID = (size_t)NANCH * 4;
static constexpr size_t SZ_HIST  = 1024 * 4;
static constexpr size_t SZ_META  = 256;
static constexpr size_t SZ_CAND  = (size_t)CAND_MAX * 8;
static constexpr size_t SZ_TOPS  = 8192;
static constexpr size_t SZ_TOPP  = 32768;
static constexpr size_t SZ_TOPV  = 8192;

static constexpr size_t OFF_FEATP = 0;
static constexpr size_t OFF_BWT   = OFF_FEATP + SZ_FEATP;
static constexpr size_t OFF_RPNB  = OFF_BWT   + SZ_BWT;
static constexpr size_t OFF_W2T   = OFF_RPNB  + SZ_RPNB;
static constexpr size_t OFF_B2    = OFF_W2T   + SZ_W2T;
static constexpr size_t OFF_OUT2  = OFF_B2    + SZ_B2;
static constexpr size_t OFF_SCORE = OFF_OUT2  + SZ_OUT2;
static constexpr size_t OFF_PROPS = OFF_SCORE + SZ_SCORE;
static constexpr size_t OFF_VALID = OFF_PROPS + SZ_PROPS;
static constexpr size_t OFF_HIST  = OFF_VALID + SZ_VALID;
static constexpr size_t OFF_META  = OFF_HIST  + SZ_HIST;
static constexpr size_t OFF_CAND  = OFF_META  + SZ_META;
static constexpr size_t OFF_TOPS  = OFF_CAND  + SZ_CAND;
static constexpr size_t OFF_TOPP  = OFF_TOPS  + SZ_TOPS;
static constexpr size_t OFF_TOPV  = OFF_TOPP  + SZ_TOPP;

// ---------------- prep kernels ----------------
// featp[c][yp][xp] (130x136, 1-px zero border, 16B-aligned rows) = feat[c][yp-1][xp-1]
__global__ void k_cvt_pad(const float* __restrict__ src, __bf16* __restrict__ dst) {
  int idx = blockIdx.x * 256 + threadIdx.x;
  if (idx >= CIN * PADH * PADW) return;
  int c   = idx / (PADH * PADW);
  int rem = idx - c * (PADH * PADW);
  int yp  = rem / PADW;
  int xp  = rem - yp * PADW;
  int y = yp - 1, x = xp - 1;
  float v = 0.f;
  if ((unsigned)y < (unsigned)HW && (unsigned)x < (unsigned)HW)
    v = src[c * (HW * HW) + y * HW + x];
  dst[idx] = (__bf16)v;
}

// Bwt[co][tap*512+ci] = conv_w[co][ci][tap]  (bf16, k contiguous per cout row)
__global__ void k_prep_w(const float* __restrict__ conv_w, __bf16* __restrict__ Bwt) {
  int idx = blockIdx.x * 256 + threadIdx.x;
  if (idx >= COUT * KTOT) return;
  int co = idx / KTOT;
  int r  = idx - co * KTOT;
  int tap = r >> 9;
  int ci  = r & 511;
  Bwt[idx] = (__bf16)conv_w[co * KTOT + ci * K3 + tap];
}

// W2t[j][k]: j=0..8 cls_w, j=9..44 box_w, 45..47 pad; b2 = combined bias
__global__ void k_prep_w2(const float* __restrict__ cls_w, const float* __restrict__ box_w,
                          const float* __restrict__ cls_b, const float* __restrict__ box_b,
                          __bf16* __restrict__ W2t, float* __restrict__ b2) {
  int idx = blockIdx.x * 256 + threadIdx.x;
  if (idx < 48 * 512) {
    int j = idx >> 9, k = idx & 511;
    float v = 0.f;
    if (j < 9)       v = cls_w[j * 512 + k];
    else if (j < 45) v = box_w[(j - 9) * 512 + k];
    W2t[idx] = (__bf16)v;
  }
  if (idx < 48) {
    float v = 0.f;
    if (idx < 9)       v = cls_b[idx];
    else if (idx < 45) v = box_b[idx - 9];
    b2[idx] = v;
  }
}

// ---------------- conv 3x3 as implicit GEMM (WMMA bf16, async-LDS, dbl-buffered) ----
// grid (128 rows, 8 cout tiles of 64), 256 threads = 8 waves.
// Outer loop over (dy, cchunk): one 130-wide A strip serves all 3 dx taps.
// B tiles (3 x 64n x 32k) copied with global_load_async_to_lds_b128 (ASYNCcnt).
__global__ __launch_bounds__(256) void k_conv3x3_wmma(
    const __bf16* __restrict__ featp,  // [512][130][136] padded
    const __bf16* __restrict__ Bwt,    // [512 cout][4608 k]
    const float*  __restrict__ bias,   // [512]
    __bf16* __restrict__ rpnb)         // [16384][512]
{
  __shared__ __align__(16) __bf16 As[2][STRIP * LDSS];
  __shared__ __align__(16) __bf16 Bs[2][3 * NT * LDSS];

  const int y     = blockIdx.x;          // image row
  const int ntile = blockIdx.y;          // cout block of 64
  const int tid  = threadIdx.x;
  const int lane = tid & 31;
  const int wave = tid >> 5;
  const int wm = wave & 3;               // 4 waves along M (32 rows each)
  const int wn = wave >> 2;              // 2 waves along N (32 cols each)
  const int half = lane >> 4;
  const int l16  = lane & 15;

  v8f acc[2][2];
#pragma unroll
  for (int f = 0; f < 2; ++f)
#pragma unroll
    for (int g = 0; g < 2; ++g) acc[f][g] = (v8f){0.f,0.f,0.f,0.f,0.f,0.f,0.f,0.f};

  auto load_tiles = [&](int it, int buf) {
    const int dy    = it >> 4;            // 0..2  (real dy-1)
    const int cbase = (it & 15) << 5;     // channel chunk * 32
    // ---- B: 3 tiles of 64 rows x 32 k, async b128 copies (one chunk per thread/tile)
    {
      const int n = tid >> 2, sub = tid & 3;
      unsigned lbase = (unsigned)(size_t)&Bs[buf][0]
                     + (unsigned)(n * LDSS * 2 + sub * 16);
      const __bf16* g0 = Bwt + (size_t)(ntile * NT + n) * KTOT
                             + dy * 3 * 512 + cbase + sub * 8;
#pragma unroll
      for (int dx = 0; dx < 3; ++dx) {
        unsigned loff = lbase + (unsigned)(dx * NT * LDSS * 2);
        unsigned long long ga = (unsigned long long)(size_t)(g0 + dx * 512);
        asm volatile("global_load_async_to_lds_b128 %0, %1, off"
                     :: "v"(loff), "v"(ga) : "memory");
      }
    }
    // ---- A: strip 136(s) x 32(k); 8-elem aligned b128 global loads, transpose into LDS
    {
      const int iy = y + dy;              // padded row index, always in [0,129]
#pragma unroll
      for (int r = 0; r < 3; ++r) {
        int chunk = r * 256 + tid;        // 544 chunks of 8 elements
        if (chunk < (PADW / 8) * 32) {
          int kk = chunk / (PADW / 8);
          int cs = (chunk - kk * (PADW / 8)) * 8;
          union { uint4 q; __bf16 e[8]; } u;
          u.q = *(const uint4*)(featp + ((size_t)(cbase + kk) * PADH + iy) * PADW + cs);
#pragma unroll
          for (int e = 0; e < 8; ++e)
            As[buf][(cs + e) * LDSS + kk] = u.e[e];
        }
      }
    }
  };

  auto compute = [&](int buf) {
    const int ka0 = half ? 8 : 0;         // A frag: K in {ka0..+7, ka0+16..+23}
    const int kb0 = half ? 8 : 0;         // B frag: K in {kb0..+7, kb0+16..+23}
#pragma unroll
    for (int dx = 0; dx < 3; ++dx) {
      v16bf af[2], bfr[2];
#pragma unroll
      for (int f = 0; f < 2; ++f) {
        int row = wm * 32 + f * 16 + l16 + dx;   // strip row = x + dx (pad-shifted)
        union { uint4 q[2]; v16bf v; } u;
        u.q[0] = *(const uint4*)&As[buf][row * LDSS + ka0];
        u.q[1] = *(const uint4*)&As[buf][row * LDSS + ka0 + 16];
        af[f] = u.v;
      }
#pragma unroll
      for (int g = 0; g < 2; ++g) {
        int col = dx * NT + wn * 32 + g * 16 + l16;
        union { uint4 q[2]; v16bf v; } u;
        u.q[0] = *(const uint4*)&Bs[buf][col * LDSS + kb0];
        u.q[1] = *(const uint4*)&Bs[buf][col * LDSS + kb0 + 16];
        bfr[g] = u.v;
      }
#pragma unroll
      for (int f = 0; f < 2; ++f)
#pragma unroll
        for (int g = 0; g < 2; ++g)
          acc[f][g] = __builtin_amdgcn_wmma_f32_16x16x32_bf16(
              false, af[f], false, bfr[g], (short)0, acc[f][g], false, false);
    }
  };

  load_tiles(0, 0);
  asm volatile("s_wait_asynccnt 0" ::: "memory");
  __syncthreads();
  int buf = 0;
  for (int it = 0; it < NK3; ++it) {
    int nxt = buf ^ 1;
    if (it + 1 < NK3) load_tiles(it + 1, nxt);   // prefetch into alternate buffer
    compute(buf);                                 // 12 WMMAs per round
    asm volatile("s_wait_asynccnt 0" ::: "memory");
    __syncthreads();
    buf = nxt;
  }

  // epilogue: bias + relu, store bf16 [pixel][cout]
#pragma unroll
  for (int f = 0; f < 2; ++f) {
#pragma unroll
    for (int g = 0; g < 2; ++g) {
      int n = ntile * NT + wn * 32 + g * 16 + l16;
      float b = bias[n];
#pragma unroll
      for (int v = 0; v < 8; ++v) {
        int m = y * HW + wm * 32 + f * 16 + v + half * 8;
        float val = fmaxf(acc[f][g][v] + b, 0.f);
        rpnb[(size_t)m * COUT + n] = (__bf16)val;
      }
    }
  }
}

// ---------------- 1x1 heads as WMMA GEMM: [16384 x 512] x [512 x 48] ----------------
__global__ __launch_bounds__(32) void k_head_wmma(
    const __bf16* __restrict__ rpnb, const __bf16* __restrict__ W2t,
    const float* __restrict__ b2, float* __restrict__ out2)
{
  const int m0   = blockIdx.x * 16;
  const int lane = threadIdx.x & 31;
  const int half = lane >> 4;
  const int l16  = lane & 15;
  v8f acc[3];
#pragma unroll
  for (int g = 0; g < 3; ++g) acc[g] = (v8f){0.f,0.f,0.f,0.f,0.f,0.f,0.f,0.f};

  for (int kt = 0; kt < 16; ++kt) {
    const int kb = kt * 32;
    v16bf a;
    {
      union { uint4 q[2]; v16bf v; } u;
      const __bf16* p = rpnb + (size_t)(m0 + l16) * 512 + kb + (half ? 8 : 0);
      u.q[0] = *(const uint4*)p;
      u.q[1] = *(const uint4*)(p + 16);
      a = u.v;
    }
#pragma unroll
    for (int g = 0; g < 3; ++g) {
      union { uint4 q[2]; v16bf v; } u;
      const __bf16* p = W2t + (size_t)(g * 16 + l16) * 512 + kb + (half ? 8 : 0);
      u.q[0] = *(const uint4*)p;
      u.q[1] = *(const uint4*)(p + 16);
      acc[g] = __builtin_amdgcn_wmma_f32_16x16x32_bf16(
          false, a, false, u.v, (short)0, acc[g], false, false);
    }
  }
#pragma unroll
  for (int g = 0; g < 3; ++g) {
    int j = g * 16 + l16;
    float b = b2[j];
#pragma unroll
    for (int v = 0; v < 8; ++v) {
      int m = m0 + v + half * 8;
      out2[(size_t)m * 48 + j] = acc[g][v] + b;
    }
  }
}

// ---------------- anchors + decode + sigmoid ----------------
__global__ void k_decode(const float* __restrict__ out2, float* __restrict__ scores,
                         float4* __restrict__ props, int* __restrict__ valid)
{
  int id = blockIdx.x * 256 + threadIdx.x;
  if (id >= NANCH) return;
  int m = id / 9, a = id - m * 9;
  int y = m >> 7, x = m & 127;

  const float sc[3] = {128.f, 256.f, 512.f};
  const float ar[3] = {0.5f, 1.f, 2.f};
  int ai = a / 3, si = a - ai * 3;
  float hr = sqrtf(ar[ai]);
  float wr = 1.f / hr;
  float wsz = wr * sc[si], hsz = hr * sc[si];
  float bx1 = rintf(-wsz * 0.5f), by1 = rintf(-hsz * 0.5f);
  float bx2 = rintf( wsz * 0.5f), by2 = rintf( hsz * 0.5f);
  float sx = (float)(x * 16), sy = (float)(y * 16);
  float ax1 = sx + bx1, ay1 = sy + by1, ax2 = sx + bx2, ay2 = sy + by2;
  float aw = ax2 - ax1, ah = ay2 - ay1;
  float acx = ax1 + 0.5f * aw, acy = ay1 + 0.5f * ah;

  const float* row = out2 + (size_t)m * 48;
  float logit = row[a];
  float score = 1.f / (1.f + expf(-logit));
  float dxv = row[9 + a * 4 + 0];
  float dyv = row[9 + a * 4 + 1];
  float dwv = fminf(row[9 + a * 4 + 2], BBOX_CLIP);
  float dhv = fminf(row[9 + a * 4 + 3], BBOX_CLIP);

  float pcx = dxv * aw + acx, pcy = dyv * ah + acy;
  float pw = expf(dwv) * aw,  ph = expf(dhv) * ah;
  float x1 = fminf(fmaxf(pcx - 0.5f * pw, 0.f), IMG_SZ);
  float y1 = fminf(fmaxf(pcy - 0.5f * ph, 0.f), IMG_SZ);
  float x2 = fminf(fmaxf(pcx + 0.5f * pw, 0.f), IMG_SZ);
  float y2 = fminf(fmaxf(pcy + 0.5f * ph, 0.f), IMG_SZ);

  scores[id] = score;
  props[id]  = make_float4(x1, y1, x2, y2);
  valid[id]  = ((x2 - x1) >= 16.f && (y2 - y1) >= 16.f) ? 1 : 0;
}

// ---------------- top-2000 selection ----------------
__global__ void k_zero(int* __restrict__ hist, int* __restrict__ meta) {
  int t = threadIdx.x;
  if (t < 1024) hist[t] = 0;
  if (t < 64)   meta[t] = 0;
}
__device__ __forceinline__ unsigned score_bin(float s) {
  unsigned b = __float_as_uint(s) >> 20;
  return b > 1023u ? 1023u : b;
}
__global__ void k_hist(const float* __restrict__ scores, int* __restrict__ hist) {
  int id = blockIdx.x * 256 + threadIdx.x;
  if (id < NANCH) atomicAdd(&hist[score_bin(scores[id])], 1);
}
__global__ void k_thresh(const int* __restrict__ hist, int* __restrict__ meta) {
  if (threadIdx.x == 0) {
    int run = 0, T = 0;
    for (int b = 1023; b >= 0; --b) { run += hist[b]; if (run >= PRE_TOPK) { T = b; break; } }
    meta[0] = T;
  }
}
__global__ void k_compact(const float* __restrict__ scores, int* __restrict__ meta,
                          unsigned long long* __restrict__ cand) {
  int id = blockIdx.x * 256 + threadIdx.x;
  if (id >= NANCH) return;
  float s = scores[id];
  if ((int)score_bin(s) >= meta[0]) {
    int pos = atomicAdd(&meta[1], 1);
    if (pos < CAND_MAX)
      cand[pos] = ((unsigned long long)__float_as_uint(s) << 32) |
                  (unsigned long long)(0xFFFFFFFFu - (unsigned)id);
  }
}
__global__ __launch_bounds__(1024) void k_sort(unsigned long long* __restrict__ cand,
                                               const int* __restrict__ meta) {
  __shared__ unsigned long long sh[CAND_MAX];
  int tid = threadIdx.x;
  int n = meta[1]; if (n > CAND_MAX) n = CAND_MAX;
  for (int i = tid; i < CAND_MAX; i += 1024) sh[i] = (i < n) ? cand[i] : 0ull;
  for (int ksz = 2; ksz <= CAND_MAX; ksz <<= 1) {
    for (int j = ksz >> 1; j > 0; j >>= 1) {
      __syncthreads();
      for (int i = tid; i < CAND_MAX; i += 1024) {
        int ixj = i ^ j;
        if (ixj > i) {
          unsigned long long a = sh[i], b = sh[ixj];
          bool desc = ((i & ksz) == 0);
          if (desc ? (a < b) : (a > b)) { sh[i] = b; sh[ixj] = a; }
        }
      }
    }
  }
  __syncthreads();
  for (int i = tid; i < CAND_MAX; i += 1024) cand[i] = sh[i];
}
__global__ void k_gather(const unsigned long long* __restrict__ cand,
                         const float4* __restrict__ props, const int* __restrict__ valid,
                         float* __restrict__ tops, float4* __restrict__ topp,
                         int* __restrict__ topv) {
  int i = blockIdx.x * 256 + threadIdx.x;
  if (i >= PRE_TOPK) return;
  unsigned long long key = cand[i];
  unsigned idx = 0xFFFFFFFFu - (unsigned)(key & 0xFFFFFFFFull);
  if (idx >= (unsigned)NANCH) idx = 0;
  tops[i] = __uint_as_float((unsigned)(key >> 32));
  topp[i] = props[idx];
  topv[i] = valid[idx];
}

// ---------------- greedy NMS + final top-1000 output ----------------
__global__ __launch_bounds__(1024) void k_nms(const float4* __restrict__ topp,
                                              const float* __restrict__ tops,
                                              const int* __restrict__ topv,
                                              float* __restrict__ out) {
  __shared__ float bx1[PRE_TOPK], by1[PRE_TOPK], bx2[PRE_TOPK], by2[PRE_TOPK], bar[PRE_TOPK];
  __shared__ int keep[PRE_TOPK];
  __shared__ int order[PRE_TOPK];
  __shared__ int cnt;
  int tid = threadIdx.x;
  for (int i = tid; i < PRE_TOPK; i += 1024) {
    float4 b = topp[i];
    bx1[i] = b.x; by1[i] = b.y; bx2[i] = b.z; by2[i] = b.w;
    bar[i] = (b.z - b.x) * (b.w - b.y);
    keep[i] = topv[i];
  }
  __syncthreads();
  for (int i = 0; i < PRE_TOPK; ++i) {
    if (keep[i]) {
      float x1i = bx1[i], y1i = by1[i], x2i = bx2[i], y2i = by2[i], ai = bar[i];
      for (int j = tid; j < PRE_TOPK; j += 1024) {
        if (j > i && keep[j]) {
          float xx1 = fmaxf(x1i, bx1[j]);
          float yy1 = fmaxf(y1i, by1[j]);
          float xx2 = fminf(x2i, bx2[j]);
          float yy2 = fminf(y2i, by2[j]);
          float inter = fmaxf(xx2 - xx1, 0.f) * fmaxf(yy2 - yy1, 0.f);
          float uni = fmaxf(ai + bar[j] - inter, 1e-9f);
          if (inter / uni > 0.7f) keep[j] = 0;
        }
      }
    }
    __syncthreads();
  }
  if (tid == 0) {
    int c = 0;
    for (int i = 0; i < PRE_TOPK; ++i) if (keep[i]) order[c++] = i;
    cnt = c;
  }
  __syncthreads();
  for (int s = tid; s < OUT_TOPK; s += 1024) {
    if (s < cnt) {
      int i = order[s];
      out[s * 4 + 0] = bx1[i];
      out[s * 4 + 1] = by1[i];
      out[s * 4 + 2] = bx2[i];
      out[s * 4 + 3] = by2[i];
      out[4 * OUT_TOPK + s] = tops[i];
    } else {
      out[s * 4 + 0] = 0.f; out[s * 4 + 1] = 0.f;
      out[s * 4 + 2] = 0.f; out[s * 4 + 3] = 0.f;
      out[4 * OUT_TOPK + s] = -1.f;
    }
  }
}

// ---------------- host launch ----------------
extern "C" void kernel_launch(void* const* d_in, const int* in_sizes, int n_in,
                              void* d_out, int out_size, void* d_ws, size_t ws_size,
                              hipStream_t stream) {
  (void)in_sizes; (void)n_in; (void)out_size; (void)ws_size;
  const float* feat   = (const float*)d_in[1];
  const float* conv_w = (const float*)d_in[2];
  const float* conv_b = (const float*)d_in[3];
  const float* cls_w  = (const float*)d_in[4];
  const float* cls_b  = (const float*)d_in[5];
  const float* box_w  = (const float*)d_in[6];
  const float* box_b  = (const float*)d_in[7];
  float* out = (float*)d_out;
  char* ws = (char*)d_ws;

  __bf16* featp = (__bf16*)(ws + OFF_FEATP);
  __bf16* Bwt   = (__bf16*)(ws + OFF_BWT);
  __bf16* rpnb  = (__bf16*)(ws + OFF_RPNB);
  __bf16* W2t   = (__bf16*)(ws + OFF_W2T);
  float*  b2    = (float*)(ws + OFF_B2);
  float*  out2  = (float*)(ws + OFF_OUT2);
  float*  scores= (float*)(ws + OFF_SCORE);
  float4* props = (float4*)(ws + OFF_PROPS);
  int*    valid = (int*)(ws + OFF_VALID);
  int*    hist  = (int*)(ws + OFF_HIST);
  int*    meta  = (int*)(ws + OFF_META);
  unsigned long long* cand = (unsigned long long*)(ws + OFF_CAND);
  float*  tops  = (float*)(ws + OFF_TOPS);
  float4* topp  = (float4*)(ws + OFF_TOPP);
  int*    topv  = (int*)(ws + OFF_TOPV);

  k_cvt_pad<<<(CIN * PADH * PADW + 255) / 256, 256, 0, stream>>>(feat, featp);
  k_prep_w<<<(COUT * KTOT + 255) / 256, 256, 0, stream>>>(conv_w, Bwt);
  k_prep_w2<<<(48 * 512 + 255) / 256, 256, 0, stream>>>(cls_w, box_w, cls_b, box_b, W2t, b2);

  dim3 gconv(HW, COUT / NT);
  k_conv3x3_wmma<<<gconv, 256, 0, stream>>>(featp, Bwt, conv_b, rpnb);
  k_head_wmma<<<MTOT / 16, 32, 0, stream>>>(rpnb, W2t, b2, out2);

  k_decode<<<(NANCH + 255) / 256, 256, 0, stream>>>(out2, scores, props, valid);
  k_zero<<<1, 1024, 0, stream>>>(hist, meta);
  k_hist<<<(NANCH + 255) / 256, 256, 0, stream>>>(scores, hist);
  k_thresh<<<1, 32, 0, stream>>>(hist, meta);
  k_compact<<<(NANCH + 255) / 256, 256, 0, stream>>>(scores, meta, cand);
  k_sort<<<1, 1024, 0, stream>>>(cand, meta);
  k_gather<<<(PRE_TOPK + 255) / 256, 256, 0, stream>>>(cand, props, valid, tops, topp, topv);
  k_nms<<<1, 1024, 0, stream>>>(topp, tops, topv, out);
}
